// GAT_88648124991292
// MI455X (gfx1250) — matrix operand order repaired and checked by reading, not compiled, not run on hardware
//
#include <hip/hip_runtime.h>

// GAT (PyG GATConv heads=8, concat=False) + global_mean_pool + linear for MI455X (gfx1250).
// GEMM h = x@W runs on v_wmma_f32_16x16x32_f16 with pre-packed f16 operands;
// edge phase exploits 192MB L2 residency of h (51.2MB).

typedef __attribute__((ext_vector_type(16))) _Float16 v16h;
typedef __attribute__((ext_vector_type(8)))  float    v8f;
typedef __attribute__((ext_vector_type(4)))  unsigned u32x4;   // one 128-bit load (8 f16)

#define IN_F 128   // input features
#define HC   256   // H*C
#define NH   8     // heads
#define CH   32    // channels per head
#define NG   64    // graphs
#define NEG_SLOPE 0.2f

// order-preserving float -> uint key (for atomicMax-based segment max)
static __device__ __forceinline__ unsigned fkey(float f) {
    unsigned b = __builtin_bit_cast(unsigned, f);
    return b ^ ((unsigned)(((int)b) >> 31) | 0x80000000u);
}
static __device__ __forceinline__ float fdec(unsigned u) {
    unsigned b = (u & 0x80000000u) ? (u ^ 0x80000000u) : ~u;
    return __builtin_bit_cast(float, b);
}

static __device__ __forceinline__ float lrelu(float v) {
    return v > 0.0f ? v : v * NEG_SLOPE;
}

// ---------------- prep A: x[N][128] f32 -> xh[N][128] f16 ----------------
__global__ void gat_xh_kernel(const float* __restrict__ x, unsigned short* __restrict__ xh,
                              long total) {
    long i = (long)blockIdx.x * blockDim.x + threadIdx.x;
    if (i >= total) return;
    xh[i] = __builtin_bit_cast(unsigned short, (_Float16)x[i]);
}

// ---------------- prep B: W[128][256] f32 -> Wt[256][128] f16 (transposed) ----------------
__global__ void gat_wt_kernel(const float* __restrict__ W, unsigned short* __restrict__ wt) {
    int i = blockIdx.x * blockDim.x + threadIdx.x;
    if (i >= IN_F * HC) return;
    int k = i / HC, n = i % HC;
    wt[n * IN_F + k] = __builtin_bit_cast(unsigned short, (_Float16)W[i]);
}

// ---------------- WMMA GEMM: h[N][256] = x[N][128] @ W ----------------
// Block = 256 thr (8 waves). Block owns 16 node-rows; wave w owns cols [32w,32w+32).
// Per k-step: 2x b128 (A) + 4x b128 (B0,B1) + 2x v_wmma; K loop fully unrolled.
__global__ void __launch_bounds__(256) gat_gemm_kernel(const unsigned short* __restrict__ xh,
                                                       const unsigned short* __restrict__ wt,
                                                       float* __restrict__ h, int N) {
    const int row0 = blockIdx.x * 16;
    const int wave = threadIdx.x >> 5;
    const int lane = threadIdx.x & 31;
    const int half = lane >> 4;
    const int m    = lane & 15;
    int row = row0 + m;
    if (row >= N) row = N - 1;            // clamp: always-valid loads; dup rows masked at store
    const int nc0 = wave * 32;

    union Frag { v16h v; u32x4 q[2]; };
    Frag A, B0, B1;
    v8f c0 = {}, c1 = {};

    const u32x4* xq  = (const u32x4*)(xh + (size_t)row * IN_F);          // unit = 8 f16
    const u32x4* b0q = (const u32x4*)(wt + (size_t)(nc0 + m) * IN_F);
    const u32x4* b1q = (const u32x4*)(wt + (size_t)(nc0 + 16 + m) * IN_F);

    #pragma unroll
    for (int s = 0; s < 4; ++s) {
        const int k0 = s * 32;
        // A: lanes 0-15 hold K=k0..k0+15 (row m), lanes 16-31 K=k0+16..k0+31 (matches B)
        const int ka = (k0 + 16 * half) >> 3;
        A.q[0] = xq[ka];
        A.q[1] = xq[ka + 1];
        // B: lanes 0-15 hold K=k0..k0+15 (col nc0+m), lanes 16-31 K=k0+16..k0+31
        const int kb = (k0 + 16 * half) >> 3;
        B0.q[0] = b0q[kb];  B0.q[1] = b0q[kb + 1];
        B1.q[0] = b1q[kb];  B1.q[1] = b1q[kb + 1];
        c0 = __builtin_amdgcn_wmma_f32_16x16x32_f16(false, A.v, false, B0.v, (short)0, c0, false, false);
        c1 = __builtin_amdgcn_wmma_f32_16x16x32_f16(false, A.v, false, B1.v, (short)0, c1, false, false);
    }
    // D layout: lane l, VGPR r -> (M = r + 8*(l>>4), N = l&15)
    #pragma unroll
    for (int r = 0; r < 8; ++r) {
        int rr = row0 + r + 8 * half;
        if (rr < N) {
            h[(size_t)rr * HC + nc0 + m]      = c0[r];
            h[(size_t)rr * HC + nc0 + 16 + m] = c1[r];
        }
    }
}

// ---------------- attention coefficients: a_src/a_dst [N][8] ----------------
// One wave per node; lane = channel; 8 wave-wide dot reductions.
__global__ void gat_att_kernel(const float* __restrict__ h,
                               const float* __restrict__ att_src,
                               const float* __restrict__ att_dst,
                               float* __restrict__ a_src, float* __restrict__ a_dst, int N) {
    int n = (blockIdx.x * blockDim.x + threadIdx.x) >> 5;
    int lane = threadIdx.x & 31;
    if (n >= N) return;
    for (int hd = 0; hd < NH; ++hd) {
        float v = h[(size_t)n * HC + hd * CH + lane];
        float p = v * att_src[hd * CH + lane];
        float q = v * att_dst[hd * CH + lane];
        #pragma unroll
        for (int msk = 16; msk >= 1; msk >>= 1) {
            p += __shfl_xor(p, msk, 32);
            q += __shfl_xor(q, msk, 32);
        }
        if (lane == 0) { a_src[n * NH + hd] = p; a_dst[n * NH + hd] = q; }
    }
}

// ---------------- edge pass 1: segment max (via ordered-uint atomicMax) ----------------
__global__ void gat_emax_kernel(const int* __restrict__ ei, int E, int EN,
                                const float* __restrict__ a_src, const float* __restrict__ a_dst,
                                unsigned* __restrict__ m_u) {
    int e = blockIdx.x * blockDim.x + threadIdx.x;
    if (e >= EN) return;
    int s, d;
    if (e < E) { s = ei[e]; d = ei[E + e]; } else { s = d = e - E; }
    #pragma unroll
    for (int hd = 0; hd < NH; ++hd) {
        float v = lrelu(a_src[s * NH + hd] + a_dst[d * NH + hd]);
        atomicMax(&m_u[d * NH + hd], fkey(v));
    }
}

// ---------------- edge pass 2: segment sum of exp(e - m[dst]) ----------------
__global__ void gat_esum_kernel(const int* __restrict__ ei, int E, int EN,
                                const float* __restrict__ a_src, const float* __restrict__ a_dst,
                                const unsigned* __restrict__ m_u, float* __restrict__ denom) {
    int e = blockIdx.x * blockDim.x + threadIdx.x;
    if (e >= EN) return;
    int s, d;
    if (e < E) { s = ei[e]; d = ei[E + e]; } else { s = d = e - E; }
    #pragma unroll
    for (int hd = 0; hd < NH; ++hd) {
        float v = lrelu(a_src[s * NH + hd] + a_dst[d * NH + hd]);
        float w = expf(v - fdec(m_u[d * NH + hd]));
        atomicAdd(&denom[d * NH + hd], w);
    }
}

// ---------------- edge pass 3: hid_acc[dst][c] += mean_h( h[src][hd][c] * alpha ) ----------------
// One wave per edge; lane = channel c; head-mean folded in (factor 1/8).
__global__ void gat_eagg_kernel(const int* __restrict__ ei, int E, int EN,
                                const float* __restrict__ h,
                                const float* __restrict__ a_src, const float* __restrict__ a_dst,
                                const unsigned* __restrict__ m_u, const float* __restrict__ denom,
                                float* __restrict__ hid_acc) {
    int e = (blockIdx.x * blockDim.x + threadIdx.x) >> 5;
    int lane = threadIdx.x & 31;
    if (e >= EN) return;
    int s, d;
    if (e < E) { s = ei[e]; d = ei[E + e]; } else { s = d = e - E; }
    float alpha = 0.0f;
    if (lane < NH) {
        float v = lrelu(a_src[s * NH + lane] + a_dst[d * NH + lane]);
        float w = expf(v - fdec(m_u[d * NH + lane]));
        alpha = w / (denom[d * NH + lane] + 1e-16f);
    }
    float acc = 0.0f;
    #pragma unroll
    for (int hd = 0; hd < NH; ++hd) {
        float al = __shfl(alpha, hd, 32);
        acc += h[(size_t)s * HC + hd * CH + lane] * al;
    }
    atomicAdd(&hid_acc[(size_t)d * CH + lane], acc * 0.125f);
}

// ---------------- pool: relu(hid+bias), mean-pool accumulation ----------------
__global__ void gat_pool_kernel(const float* __restrict__ hid_acc, const float* __restrict__ bias,
                                const int* __restrict__ batch,
                                float* __restrict__ psum, float* __restrict__ cnt, int N) {
    int n = (blockIdx.x * blockDim.x + threadIdx.x) >> 5;
    int lane = threadIdx.x & 31;
    if (n >= N) return;
    float v = fmaxf(hid_acc[(size_t)n * CH + lane] + bias[lane], 0.0f);
    int g = batch[n];
    atomicAdd(&psum[g * CH + lane], v);
    if (lane == 0) atomicAdd(&cnt[g], 1.0f);
}

// ---------------- head: pooled = sum/cnt ; x_t = pooled@lin_w + lin_b ----------------
// d_out = [x_t (64*2) | pooled (64*32)]
__global__ void gat_final_kernel(const float* __restrict__ psum, const float* __restrict__ cnt,
                                 const float* __restrict__ lin_w, const float* __restrict__ lin_b,
                                 float* __restrict__ out) {
    int g = threadIdx.x;
    if (g >= NG) return;
    float cg = fmaxf(cnt[g], 1.0f);
    float acc0 = lin_b[0], acc1 = lin_b[1];
    for (int c = 0; c < CH; ++c) {
        float p = psum[g * CH + c] / cg;
        out[NG * 2 + g * CH + c] = p;
        acc0 += p * lin_w[c * 2 + 0];
        acc1 += p * lin_w[c * 2 + 1];
    }
    out[g * 2 + 0] = acc0;
    out[g * 2 + 1] = acc1;
}

extern "C" void kernel_launch(void* const* d_in, const int* in_sizes, int n_in,
                              void* d_out, int out_size, void* d_ws, size_t ws_size,
                              hipStream_t stream) {
    const float* x       = (const float*)d_in[0];
    const int*   ei      = (const int*)d_in[1];
    const int*   batch   = (const int*)d_in[2];
    const float* W       = (const float*)d_in[3];
    const float* att_src = (const float*)d_in[4];
    const float* att_dst = (const float*)d_in[5];
    const float* bias    = (const float*)d_in[6];
    const float* lin_w   = (const float*)d_in[7];
    const float* lin_b   = (const float*)d_in[8];
    (void)n_in; (void)out_size; (void)ws_size;

    const int N  = in_sizes[0] / IN_F;
    const int E  = in_sizes[1] / 2;
    const int EN = E + N;

    // workspace carve-out (256B aligned)
    char* ws = (char*)d_ws;
    size_t off = 0;
    auto carve = [&](size_t bytes) -> void* {
        void* p = ws + off;
        off = (off + bytes + 255) & ~(size_t)255;
        return p;
    };
    unsigned short* xh    = (unsigned short*)carve((size_t)N * IN_F * 2);
    unsigned short* wt    = (unsigned short*)carve((size_t)IN_F * HC * 2);
    float*          h     = (float*)carve((size_t)N * HC * 4);
    float*          asrcb = (float*)carve((size_t)N * NH * 4);
    float*          adstb = (float*)carve((size_t)N * NH * 4);
    unsigned*       m_u   = (unsigned*)carve((size_t)N * NH * 4);
    float*          denom = (float*)carve((size_t)N * NH * 4);
    float*          hid   = (float*)carve((size_t)N * CH * 4);
    float*          psum  = (float*)carve((size_t)NG * CH * 4);
    float*          cnt   = (float*)carve((size_t)NG * 4);

    // zero-init (m_u key 0 == "-inf" sentinel under the order-preserving map)
    hipMemsetAsync(m_u,   0, (size_t)N * NH * 4, stream);
    hipMemsetAsync(denom, 0, (size_t)N * NH * 4, stream);
    hipMemsetAsync(hid,   0, (size_t)N * CH * 4, stream);
    hipMemsetAsync(psum,  0, (size_t)NG * CH * 4, stream);
    hipMemsetAsync(cnt,   0, (size_t)NG * 4, stream);

    const long xtot = (long)N * IN_F;
    gat_xh_kernel<<<(unsigned)((xtot + 255) / 256), 256, 0, stream>>>(x, xh, xtot);
    gat_wt_kernel<<<(IN_F * HC + 255) / 256, 256, 0, stream>>>(W, wt);
    gat_gemm_kernel<<<(N + 15) / 16, 256, 0, stream>>>(xh, wt, h, N);
    gat_att_kernel<<<((size_t)N * 32 + 255) / 256, 256, 0, stream>>>(h, att_src, att_dst, asrcb, adstb, N);
    gat_emax_kernel<<<(EN + 255) / 256, 256, 0, stream>>>(ei, E, EN, asrcb, adstb, m_u);
    gat_esum_kernel<<<(EN + 255) / 256, 256, 0, stream>>>(ei, E, EN, asrcb, adstb, m_u, denom);
    gat_eagg_kernel<<<((size_t)EN * 32 + 255) / 256, 256, 0, stream>>>(ei, E, EN, h, asrcb, adstb, m_u, denom, hid);
    gat_pool_kernel<<<((size_t)N * 32 + 255) / 256, 256, 0, stream>>>(hid, bias, batch, psum, cnt, N);
    gat_final_kernel<<<1, 64, 0, stream>>>(psum, cnt, lin_w, lin_b, (float*)d_out);
}
